// PraxisMemory_28862180229127
// MI455X (gfx1250) — matrix operand order repaired and checked by reading, not compiled, not run on hardware
//
#include <hip/hip_runtime.h>
#include <hip/hip_bf16.h>

typedef __attribute__((ext_vector_type(16))) __bf16 v16bf;
typedef __attribute__((ext_vector_type(8)))  float  v8f;

#define PM_EPS 1e-8f

constexpr int B = 4, H = 16, S = 4096, D = 64;
constexpr int CH = 128;              // tokens per chunk
constexpr int NCHUNK = S / CH;       // 32
constexpr int SEG = 8;               // S-segments per (b,h) for parallel reduction
constexpr int CPG = NCHUNK / SEG;    // 4 chunks per segment

__device__ __forceinline__ float elu1(float x) { return x > 0.f ? x + 1.f : __expf(x); }

// A fragment (16x32 bf16, row-major source, leading dim ld):
// lanes 0-15: m=lane, K = k0+0..7 and k0+16..23
// lanes 16-31: m=lane-16, K = k0+8..15 and k0+24..31
__device__ __forceinline__ v16bf load_a_frag(const __bf16* m, int r0, int k0, int ld, int lane) {
  const int lo = lane & 15, hi = lane >> 4;
  const __bf16* p = m + (r0 + lo) * ld + k0 + hi * 8;
  v16bf a;
#pragma unroll
  for (int i = 0; i < 8; ++i) { a[i] = p[i]; a[8 + i] = p[16 + i]; }
  return a;
}

// B fragment (32x16 bf16) from TRANSPOSED storage Bt[n*ld + k] == B[k][n]:
// lanes 0-15 hold K=k0+0..7 and k0+16..23 for n=lane; lanes 16-31 hold K=k0+8..15 and k0+24..31.
__device__ __forceinline__ v16bf load_b_frag_t(const __bf16* mt, int n0, int k0, int ld, int lane) {
  const int lo = lane & 15, hi = lane >> 4;
  const __bf16* p = mt + (n0 + lo) * ld + k0 + hi * 8;
  v16bf b;
#pragma unroll
  for (int i = 0; i < 8; ++i) { b[i] = p[i]; b[8 + i] = p[16 + i]; }
  return b;
}

__device__ __forceinline__ v8f wmma_bf16(v16bf a, v16bf b, v8f c) {
  return __builtin_amdgcn_wmma_f32_16x16x32_bf16(false, a, false, b, (short)0, c, false, false);
}

// ---------------------------------------------------------------------------
// Kernel A: one WG per (b,h,segment) -> 512 WGs. Streams its 512 tokens in
// 128-token chunks: sigma_k -> GEMM1 (num = Sk @ W) -> value_update ->
// GEMM2 (delta += Sk^T @ Vu). Delta lives in per-wave WMMA accumulators for
// the whole segment; partial delta / partial colsum written to workspace.
// Deterministic (no atomics); reduction order fixed by kernel B.
// ---------------------------------------------------------------------------
__global__ void __launch_bounds__(256) praxis_update_partial(
    const float* __restrict__ key, const float* __restrict__ value,
    const float* __restrict__ init_states, const float* __restrict__ init_z,
    float* __restrict__ ws_pdelta, float* __restrict__ ws_pz) {
  __shared__ __bf16 Wt[D * D];        // Wt[e*64+d] = W[d][e]  (B operand, GEMM1)
  __shared__ __bf16 Sk[CH * D];       // [s][d]                (A operand, GEMM1)
  __shared__ __bf16 SkT[D * CH];      // [d][s]                (A operand, GEMM2)
  __shared__ __bf16 VuT[D * CH];      // [e][s]                (B operand, GEMM2)
  __shared__ float  zsh[D];
  __shared__ float  densh[CH];

  const int blk = blockIdx.x;
  const int bh  = blk / SEG;
  const int seg = blk - bh * SEG;
  const int h   = bh & (H - 1);
  const int t   = threadIdx.x;
  const int wv  = t >> 5, lane = t & 31;
  const int lo  = lane & 15, hi = lane >> 4;

  const size_t kvbase = (size_t)bh * S * D;
  const size_t wbase  = (size_t)h * D * D;

  for (int i = t; i < D * D; i += 256) {
    int d = i >> 6, e = i & 63;
    Wt[e * D + d] = (__bf16)init_states[wbase + i];
  }
  if (t < D) zsh[t] = init_z[h * D + t];
  __syncthreads();

  float colAcc = 0.f;                           // per-thread (t<64): sum_s sigma_k[s,t]
  v8f dacc[2] = {{0,0,0,0,0,0,0,0}, {0,0,0,0,0,0,0,0}};

  const int cBeg = seg * CPG, cEnd = cBeg + CPG;
  for (int c = cBeg; c < cEnd; ++c) {
    const int s0 = c * CH;

    // ---- stage sigma_k = elu(K)+1 (bf16, both layouts) + den = sk.z + eps ----
    if (t < CH) {
      const int s = s0 + t;
      float sk[D];
      const float4* kp = (const float4*)(key + kvbase + (size_t)s * D);
      float den = PM_EPS;
#pragma unroll
      for (int i = 0; i < D / 4; ++i) {
        float4 kv = kp[i];
        float a0 = elu1(kv.x), a1 = elu1(kv.y), a2 = elu1(kv.z), a3 = elu1(kv.w);
        sk[4*i+0] = a0; sk[4*i+1] = a1; sk[4*i+2] = a2; sk[4*i+3] = a3;
        den += a0 * zsh[4*i+0] + a1 * zsh[4*i+1] + a2 * zsh[4*i+2] + a3 * zsh[4*i+3];
      }
      densh[t] = den;
#pragma unroll
      for (int d = 0; d < D; ++d) {
        __bf16 v = (__bf16)sk[d];
        Sk[t * D + d]   = v;
        SkT[d * CH + t] = v;
      }
      // prefetch next chunk's K/V rows (global_prefetch_b8) while we compute
      if (c + 1 < cEnd) {
        const float* nk = key   + kvbase + (size_t)(s + CH) * D;
        const float* nv = value + kvbase + (size_t)(s + CH) * D;
        __builtin_prefetch(nk, 0, 1);  __builtin_prefetch(nk + 32, 0, 1);
        __builtin_prefetch(nv, 0, 1);  __builtin_prefetch(nv + 32, 0, 1);
      }
    }
    __syncthreads();

    // ---- GEMM1: num = Sk(128x64) @ W(64x64); Vu = V - num/den (transposed bf16) ----
#pragma unroll
    for (int i = 0; i < 4; ++i) {
      const int tid = wv * 4 + i;
      const int mt = tid >> 2, nt = tid & 3;
      v8f acc = {0,0,0,0,0,0,0,0};
#pragma unroll
      for (int k0 = 0; k0 < D; k0 += 32) {
        v16bf a = load_a_frag(Sk, mt * 16, k0, D, lane);
        v16bf b = load_b_frag_t(Wt, nt * 16, k0, D, lane);
        acc = wmma_bf16(a, b, acc);
      }
      const int e  = nt * 16 + lo;
      const int rb = mt * 16 + hi * 8;
#pragma unroll
      for (int j = 0; j < 8; ++j) {
        const int r = rb + j;
        float vu = value[kvbase + (size_t)(s0 + r) * D + e] - acc[j] / densh[r];
        VuT[e * CH + r] = (__bf16)vu;
      }
    }
    // column sums of sigma_k for memory_z (contiguous reads of SkT row t)
    if (t < D) {
#pragma unroll 8
      for (int r = 0; r < CH; ++r) colAcc += (float)SkT[t * CH + r];
    }
    __syncthreads();

    // ---- GEMM2: delta(64x64) += Sk^T(64x128) @ Vu(128x64), K-loop over chunk ----
#pragma unroll
    for (int k0 = 0; k0 < CH; k0 += 32) {
#pragma unroll
      for (int i = 0; i < 2; ++i) {
        const int tid = wv * 2 + i;
        const int dt = tid >> 2, et = tid & 3;
        v16bf a = load_a_frag(SkT, dt * 16, k0, CH, lane);
        v16bf b = load_b_frag_t(VuT, et * 16, k0, CH, lane);
        dacc[i] = wmma_bf16(a, b, dacc[i]);
      }
    }
    __syncthreads();   // protect Sk/SkT/VuT before next chunk's stage
  }

  // ---- write this segment's partial delta / partial column sums ----
  float* pd = ws_pdelta + (size_t)blk * D * D;
#pragma unroll
  for (int i = 0; i < 2; ++i) {
    const int tid = wv * 2 + i;
    const int dt = tid >> 2, et = tid & 3;
#pragma unroll
    for (int j = 0; j < 8; ++j) {
      const int d = dt * 16 + hi * 8 + j;
      const int e = et * 16 + lo;
      pd[d * D + e] = dacc[i][j];
    }
  }
  if (t < D) ws_pz[blk * D + t] = colAcc;
}

// ---------------------------------------------------------------------------
// Kernel B: deterministic reduction of SEG partials per (b,h):
//   memory_states = init_states + sum(partials); memory_z = init_z + sum(pz)
// ---------------------------------------------------------------------------
__global__ void __launch_bounds__(256) praxis_reduce(
    const float* __restrict__ init_states, const float* __restrict__ init_z,
    const float* __restrict__ ws_pdelta, const float* __restrict__ ws_pz,
    float* __restrict__ ws_states, float* __restrict__ ws_z) {
  const int bh = blockIdx.x;
  const int h  = bh & (H - 1);
  const int t  = threadIdx.x;
  const size_t wbase = (size_t)h * D * D;
  const float* pd = ws_pdelta + (size_t)bh * SEG * D * D;

  for (int i = t; i < D * D; i += 256) {
    float s = init_states[wbase + i];
#pragma unroll
    for (int g = 0; g < SEG; ++g) s += pd[(size_t)g * D * D + i];
    ws_states[(size_t)bh * D * D + i] = s;
  }
  if (t < D) {
    float s = init_z[h * D + t];
#pragma unroll
    for (int g = 0; g < SEG; ++g) s += ws_pz[(bh * SEG + g) * D + t];
    ws_z[bh * D + t] = s;
  }
}

// ---------------------------------------------------------------------------
// Kernel C: fully parallel over (b,h,chunk): ret = sigma_q @ memory_states,
// norm = sigma_q . memory_z + eps, sigmoid-gated blend, write output.
// ---------------------------------------------------------------------------
__global__ void __launch_bounds__(256) praxis_retrieve(
    const float* __restrict__ query, const float* __restrict__ outin,
    const float* __restrict__ betas,
    const float* __restrict__ ws_states, const float* __restrict__ ws_z,
    float* __restrict__ out) {
  __shared__ __bf16 Mt[D * D];        // Mt[e*64+d] = memory_states[d][e]
  __shared__ __bf16 Sq[CH * D];       // [s][d]
  __shared__ float  zsh[D], gsh[D], densh[CH];

  const int bh = blockIdx.x >> 5;     // NCHUNK == 32
  const int c  = blockIdx.x & (NCHUNK - 1);
  const int h  = bh & (H - 1);
  const int t  = threadIdx.x;
  const int wv = t >> 5, lane = t & 31;
  const int lo = lane & 15, hi = lane >> 4;
  const int s0 = c * CH;
  const size_t base = (size_t)bh * S * D;

  for (int i = t; i < D * D; i += 256) {
    int d = i >> 6, e = i & 63;
    Mt[e * D + d] = (__bf16)ws_states[(size_t)bh * D * D + i];
  }
  if (t < D) {
    zsh[t] = ws_z[bh * D + t];
    float bv = betas[h * D + t];
    gsh[t] = 1.f / (1.f + __expf(-bv));
  }
  __syncthreads();

  if (t < CH) {
    const int s = s0 + t;
    const float4* qp = (const float4*)(query + base + (size_t)s * D);
    float den = PM_EPS;
#pragma unroll
    for (int i = 0; i < D / 4; ++i) {
      float4 qv = qp[i];
      float a0 = elu1(qv.x), a1 = elu1(qv.y), a2 = elu1(qv.z), a3 = elu1(qv.w);
      den += a0 * zsh[4*i+0] + a1 * zsh[4*i+1] + a2 * zsh[4*i+2] + a3 * zsh[4*i+3];
      Sq[t * D + 4*i+0] = (__bf16)a0; Sq[t * D + 4*i+1] = (__bf16)a1;
      Sq[t * D + 4*i+2] = (__bf16)a2; Sq[t * D + 4*i+3] = (__bf16)a3;
    }
    densh[t] = den;
  }
  __syncthreads();

#pragma unroll
  for (int i = 0; i < 4; ++i) {
    const int tid = wv * 4 + i;
    const int mt = tid >> 2, nt = tid & 3;
    v8f acc = {0,0,0,0,0,0,0,0};
#pragma unroll
    for (int k0 = 0; k0 < D; k0 += 32) {
      v16bf a = load_a_frag(Sq, mt * 16, k0, D, lane);
      v16bf b = load_b_frag_t(Mt, nt * 16, k0, D, lane);
      acc = wmma_bf16(a, b, acc);
    }
    const int e  = nt * 16 + lo;
    const int rb = mt * 16 + hi * 8;
    const float g = gsh[e];
#pragma unroll
    for (int j = 0; j < 8; ++j) {
      const int r = rb + j;
      const size_t idx = base + (size_t)(s0 + r) * D + e;
      float mo = acc[j] / densh[r];
      out[idx] = g * mo + (1.f - g) * outin[idx];
    }
  }
}

extern "C" void kernel_launch(void* const* d_in, const int* in_sizes, int n_in,
                              void* d_out, int out_size, void* d_ws, size_t ws_size,
                              hipStream_t stream) {
  (void)in_sizes; (void)n_in; (void)out_size; (void)ws_size;
  const float* query       = (const float*)d_in[0];
  const float* key         = (const float*)d_in[1];
  const float* value       = (const float*)d_in[2];
  const float* outin       = (const float*)d_in[3];
  const float* betas       = (const float*)d_in[4];
  const float* init_states = (const float*)d_in[5];
  const float* init_z      = (const float*)d_in[6];

  // Workspace layout (floats): states 1MB | z 16KB | partial deltas 8MB | partial z 128KB
  float* ws_states = (float*)d_ws;
  float* ws_z      = ws_states + (size_t)B * H * D * D;
  float* ws_pd     = ws_z + (size_t)B * H * D;
  float* ws_pz     = ws_pd + (size_t)B * H * SEG * D * D;

  praxis_update_partial<<<B * H * SEG, 256, 0, stream>>>(key, value, init_states, init_z,
                                                         ws_pd, ws_pz);
  praxis_reduce<<<B * H, 256, 0, stream>>>(init_states, init_z, ws_pd, ws_pz,
                                           ws_states, ws_z);
  praxis_retrieve<<<B * H * NCHUNK, 256, 0, stream>>>(query, outin, betas,
                                                      ws_states, ws_z, (float*)d_out);
}
